// LstmAutoencoderPk_29781303230686
// MI455X (gfx1250) — matrix operand order repaired and checked by reading, not compiled, not run on hardware
//
#include <hip/hip_runtime.h>
#include <cstddef>

typedef __bf16 bf16_t;
typedef __attribute__((ext_vector_type(8)))  __bf16 v8bf;
typedef __attribute__((ext_vector_type(16))) __bf16 v16bf;
typedef __attribute__((ext_vector_type(8)))  float  v8f;

#define BATCH 64
#define TLEN  512
#define DIN   128
#define HID   512
#define G4    2048      // 4*H gate columns
#define NWG   16        // workgroups cooperating per LSTM
#define COLS  128       // gate columns per workgroup (32 per gate)
#define LDW   648       // padded K stride (bf16 halves) for W in LDS (K=640 + pad)
#define GLD   132       // padded gate row stride (floats)
#define DSTEPS 511      // decoder steps (T-1)

union ABu { v16bf v; v8bf h[2]; };

__device__ inline v8f wmma_bf16f32(v16bf a, v16bf b, v8f c) {
  return __builtin_amdgcn_wmma_f32_16x16x32_bf16(false, a, false, b, (short)0, c,
                                                 false, false);
}

__device__ inline bf16_t f2bf(float f) {
  unsigned u = __float_as_uint(f);
  u = (u + 0x7FFFu + ((u >> 16) & 1u)) >> 16;   // round-to-nearest-even
  unsigned short s = (unsigned short)u;
  bf16_t r;
  __builtin_memcpy(&r, &s, 2);
  return r;
}

__device__ inline float sigm(float x) { return 1.f / (1.f + __expf(-x)); }

// Device-wide split barrier over the NWG workgroups of one LSTM.
__device__ inline void grid_barrier(unsigned int* cnt, unsigned int target) {
  __threadfence();            // release h stores to agent scope
  __syncthreads();
  if (threadIdx.x == 0) {
    __hip_atomic_fetch_add(cnt, 1u, __ATOMIC_ACQ_REL, __HIP_MEMORY_SCOPE_AGENT);
    while (__hip_atomic_load(cnt, __ATOMIC_ACQUIRE, __HIP_MEMORY_SCOPE_AGENT) < target) {
      __builtin_amdgcn_s_sleep(2);
    }
  }
  __syncthreads();
  __threadfence();            // acquire: invalidate stale WGP$ lines before h reloads
}

// ---------------------------------------------------------------------------
// small helpers
// ---------------------------------------------------------------------------
__global__ void zero_u32(unsigned int* p, int n) {
  int i = blockIdx.x * blockDim.x + threadIdx.x;
  if (i < n) p[i] = 0u;
}

__global__ void cvt_kernel(bf16_t* dst, const float* src, int n) {
  int i = blockIdx.x * blockDim.x + threadIdx.x;
  if (i < n) dst[i] = f2bf(src[i]);
}

// ---------------------------------------------------------------------------
// Encoder: forward LSTM (blocks 0..15) + backward LSTM (blocks 16..31)
// Each block owns 32 h-columns; per step computes gates = [h, x_t] @ Wl^T via
// WMMA with Wl ([W_hh;W_ih] chunk, 640x128) cached in LDS.  Accumulates masked
// mean and final h directly into the latent-input buffer.
// ---------------------------------------------------------------------------
__global__ __launch_bounds__(256) void lstm_encoder_kernel(
    const bf16_t* __restrict__ xbf, const int* __restrict__ lengths,
    const bf16_t* __restrict__ whh_all, const bf16_t* __restrict__ wih_all,
    const float* __restrict__ bias_f, const float* __restrict__ bias_b,
    bf16_t* __restrict__ hbuf,      // [2 lstm][2 parity][64][512]
    float* __restrict__ latentin,   // [64][2048] = [hf | hb | mean_f | mean_b]
    unsigned int* __restrict__ counters)
{
  extern __shared__ char smem[];
  bf16_t* Wl  = (bf16_t*)smem;                               // [COLS][LDW]
  float*  gat = (float*)(smem + COLS * LDW * sizeof(bf16_t)); // [64][GLD]
  float*  bl  = gat + BATCH * GLD;                            // [COLS]

  const int lstm = (int)blockIdx.x >> 4;    // 0 = forward, 1 = backward
  const int wg   = (int)blockIdx.x & 15;
  const int tid  = threadIdx.x;
  const int lane = tid & 31;
  const int wave = tid >> 5;
  const int jbase = wg * 32;

  const bf16_t* whh = whh_all + (size_t)lstm * G4 * HID;
  const bf16_t* wih = wih_all + (size_t)lstm * G4 * DIN;
  const float* bias = lstm ? bias_b : bias_f;
  bf16_t* hb = hbuf + (size_t)lstm * 2 * BATCH * HID;
  unsigned int* cnt = counters + lstm * 16;

  // Stage fused weight chunk [W_hh | W_ih] for our 128 gate columns into LDS.
  for (int idx = tid; idx < COLS * 80; idx += 256) {
    int c = idx / 80, ch = idx % 80;
    int g = ((c >> 5) * HID) + jbase + (c & 31);
    v8bf v;
    if (ch < 64) v = *(const v8bf*)(whh + (size_t)g * HID + ch * 8);
    else         v = *(const v8bf*)(wih + (size_t)g * DIN + (ch - 64) * 8);
    *(v8bf*)(Wl + c * LDW + ch * 8) = v;
  }
  if (tid < COLS) bl[tid] = bias[((tid >> 5) * HID) + jbase + (tid & 31)];
  __syncthreads();

  // Elementwise-phase ownership: 8 consecutive j per thread, fixed batch row.
  const int eb  = tid >> 2;
  const int ej0 = (tid & 3) * 8;
  const int ilen = lengths[eb];
  const float flen = (float)ilen;
  float creg[8], hsum[8], hreg[8];
#pragma unroll
  for (int i = 0; i < 8; ++i) { creg[i] = 0.f; hsum[i] = 0.f; hreg[i] = 0.f; }

  // WMMA tile assignment: wave -> (M-tile, 4 N-tiles)
  const int mt   = wave & 3;
  const int nh   = wave >> 2;
  const int arow = mt * 16 + (lane & 15);
  const int koff = (lane >> 4) << 3;   // A: 8-granule per lane half
  const int kb   = (lane >> 4) << 3;   // B: 8-granule per lane half

  for (int s = 0; s < TLEN; ++s) {
    const int tt = lstm ? (TLEN - 1 - s) : s;
    const bf16_t* hprev = hb + (size_t)(s & 1) * BATCH * HID;
    bf16_t* hnext = hb + (size_t)((s + 1) & 1) * BATCH * HID;

    v8f acc[4];
#pragma unroll
    for (int n = 0; n < 4; ++n)
#pragma unroll
      for (int i = 0; i < 8; ++i) acc[n][i] = 0.f;

    for (int kit = 0; kit < 20; ++kit) {
      const int base = kit * 32 + koff;
      ABu a;
      const bf16_t* ap = (kit < 16)
          ? (hprev + (size_t)arow * HID + base)
          : (xbf + ((size_t)arow * TLEN + tt) * DIN + (base - HID));
      a.h[0] = *(const v8bf*)ap;
      a.h[1] = *(const v8bf*)(ap + 16);
      const int baseB = kit * 32 + kb;
#pragma unroll
      for (int n = 0; n < 4; ++n) {
        const int col = nh * 64 + n * 16 + (lane & 15);
        ABu bm;
        const bf16_t* bp = Wl + col * LDW + baseB;
        bm.h[0] = *(const v8bf*)bp;
        bm.h[1] = *(const v8bf*)(bp + 16);
        acc[n] = wmma_bf16f32(a.v, bm.v, acc[n]);
      }
    }
    // Scatter C (f32, ISA C/D layout) to gate staging in LDS.
#pragma unroll
    for (int n = 0; n < 4; ++n) {
      const int col = nh * 64 + n * 16 + (lane & 15);
#pragma unroll
      for (int v = 0; v < 8; ++v) {
        const int row = mt * 16 + v + ((lane >> 4) << 3);
        gat[row * GLD + col] = acc[n][v];
      }
    }
    __syncthreads();

    // Elementwise LSTM cell.
    v8bf hv;
#pragma unroll
    for (int i = 0; i < 8; ++i) {
      const int j = ej0 + i;
      const float gi = gat[eb * GLD + j]      + bl[j];
      const float gf = gat[eb * GLD + 32 + j] + bl[32 + j];
      const float gg = gat[eb * GLD + 64 + j] + bl[64 + j];
      const float go = gat[eb * GLD + 96 + j] + bl[96 + j];
      const float c = sigm(gf) * creg[i] + sigm(gi) * tanhf(gg);
      creg[i] = c;
      const float h = sigm(go) * tanhf(c);
      hreg[i] = h;
      if (tt < ilen) hsum[i] += h;
      hv[i] = f2bf(h);
    }
    *(v8bf*)(hnext + (size_t)eb * HID + jbase + ej0) = hv;

    if (s < TLEN - 1) grid_barrier(cnt, (unsigned)(s + 1) * NWG);
  }

  // Final h and masked mean into latent input.
  const int hoff = lstm ? HID : 0;
  const int moff = lstm ? (3 * HID) : (2 * HID);
#pragma unroll
  for (int i = 0; i < 8; ++i) {
    const int jg = jbase + ej0 + i;
    latentin[(size_t)eb * G4 + hoff + jg] = hreg[i];
    latentin[(size_t)eb * G4 + moff + jg] = hsum[i] / flen;
  }
}

// ---------------------------------------------------------------------------
// Latent: latent = latentin @ W_lat^T + b_lat ; h_dec = latent @ W_l2h^T + b_l2h
// Small (~42 MFLOP) — one block, f32 VALU.
// ---------------------------------------------------------------------------
__global__ __launch_bounds__(256) void latent_kernel(
    const float* __restrict__ latentin, const float* __restrict__ W_lat,
    const float* __restrict__ b_lat, const float* __restrict__ W_l2h,
    const float* __restrict__ b_l2h, bf16_t* __restrict__ hdec)
{
  __shared__ float lat[BATCH * 128];
  const int tid = threadIdx.x;
  for (int o = tid; o < BATCH * 128; o += 256) {
    const int b = o >> 7, l = o & 127;
    const float* xr = latentin + (size_t)b * G4;
    const float* wr = W_lat + (size_t)l * G4;
    float acc = b_lat[l];
    for (int k = 0; k < G4; ++k) acc = fmaf(xr[k], wr[k], acc);
    lat[o] = acc;
  }
  __syncthreads();
  for (int o = tid; o < BATCH * HID; o += 256) {
    const int b = o >> 9, j = o & 511;
    const float* wr = W_l2h + (size_t)j * 128;
    const float* lr = lat + b * 128;
    float acc = b_l2h[j];
    for (int k = 0; k < 128; ++k) acc = fmaf(lr[k], wr[k], acc);
    hdec[o] = f2bf(acc);
  }
}

// ---------------------------------------------------------------------------
// Decoder LSTM: h0 = h_dec, c0 = 0, 511 steps; writes y history (bf16) which
// doubles as the previous-h source for the next step.
// ---------------------------------------------------------------------------
__global__ __launch_bounds__(256) void lstm_decoder_kernel(
    const bf16_t* __restrict__ xbf,
    const bf16_t* __restrict__ whh, const bf16_t* __restrict__ wih,
    const float* __restrict__ bias,
    const bf16_t* __restrict__ hdec, bf16_t* __restrict__ yhist,
    unsigned int* __restrict__ cnt)
{
  extern __shared__ char smem[];
  bf16_t* Wl  = (bf16_t*)smem;
  float*  gat = (float*)(smem + COLS * LDW * sizeof(bf16_t));
  float*  bl  = gat + BATCH * GLD;

  const int wg   = (int)blockIdx.x;
  const int tid  = threadIdx.x;
  const int lane = tid & 31;
  const int wave = tid >> 5;
  const int jbase = wg * 32;

  for (int idx = tid; idx < COLS * 80; idx += 256) {
    int c = idx / 80, ch = idx % 80;
    int g = ((c >> 5) * HID) + jbase + (c & 31);
    v8bf v;
    if (ch < 64) v = *(const v8bf*)(whh + (size_t)g * HID + ch * 8);
    else         v = *(const v8bf*)(wih + (size_t)g * DIN + (ch - 64) * 8);
    *(v8bf*)(Wl + c * LDW + ch * 8) = v;
  }
  if (tid < COLS) bl[tid] = bias[((tid >> 5) * HID) + jbase + (tid & 31)];
  __syncthreads();

  const int eb  = tid >> 2;
  const int ej0 = (tid & 3) * 8;
  float creg[8];
#pragma unroll
  for (int i = 0; i < 8; ++i) creg[i] = 0.f;

  const int mt   = wave & 3;
  const int nh   = wave >> 2;
  const int arow = mt * 16 + (lane & 15);
  const int koff = (lane >> 4) << 3;
  const int kb   = (lane >> 4) << 3;

  for (int s = 0; s < DSTEPS; ++s) {
    const bf16_t* hprev =
        (s == 0) ? hdec : (yhist + (size_t)(s - 1) * BATCH * HID);
    bf16_t* hnext = yhist + (size_t)s * BATCH * HID;

    v8f acc[4];
#pragma unroll
    for (int n = 0; n < 4; ++n)
#pragma unroll
      for (int i = 0; i < 8; ++i) acc[n][i] = 0.f;

    for (int kit = 0; kit < 20; ++kit) {
      const int base = kit * 32 + koff;
      ABu a;
      const bf16_t* ap = (kit < 16)
          ? (hprev + (size_t)arow * HID + base)
          : (xbf + ((size_t)arow * TLEN + s) * DIN + (base - HID));
      a.h[0] = *(const v8bf*)ap;
      a.h[1] = *(const v8bf*)(ap + 16);
      const int baseB = kit * 32 + kb;
#pragma unroll
      for (int n = 0; n < 4; ++n) {
        const int col = nh * 64 + n * 16 + (lane & 15);
        ABu bm;
        const bf16_t* bp = Wl + col * LDW + baseB;
        bm.h[0] = *(const v8bf*)bp;
        bm.h[1] = *(const v8bf*)(bp + 16);
        acc[n] = wmma_bf16f32(a.v, bm.v, acc[n]);
      }
    }
#pragma unroll
    for (int n = 0; n < 4; ++n) {
      const int col = nh * 64 + n * 16 + (lane & 15);
#pragma unroll
      for (int v = 0; v < 8; ++v) {
        const int row = mt * 16 + v + ((lane >> 4) << 3);
        gat[row * GLD + col] = acc[n][v];
      }
    }
    __syncthreads();

    v8bf hv;
#pragma unroll
    for (int i = 0; i < 8; ++i) {
      const int j = ej0 + i;
      const float gi = gat[eb * GLD + j]      + bl[j];
      const float gf = gat[eb * GLD + 32 + j] + bl[32 + j];
      const float gg = gat[eb * GLD + 64 + j] + bl[64 + j];
      const float go = gat[eb * GLD + 96 + j] + bl[96 + j];
      const float c = sigm(gf) * creg[i] + sigm(gi) * tanhf(gg);
      creg[i] = c;
      hv[i] = f2bf(sigm(go) * tanhf(c));
    }
    *(v8bf*)(hnext + (size_t)eb * HID + jbase + ej0) = hv;

    if (s < DSTEPS - 1) grid_barrier(cnt, (unsigned)(s + 1) * NWG);
  }
}

// ---------------------------------------------------------------------------
// Output projection + skip:  out = a*(y @ W_out^T + b_out) + (1-a)*x[:,1:,:]
// M = 511*64 rows (t-major), N = 128, K = 512; WMMA.
// ---------------------------------------------------------------------------
__global__ __launch_bounds__(256) void out_proj_kernel(
    const bf16_t* __restrict__ yhist, const bf16_t* __restrict__ woutb,
    const float* __restrict__ b_out, const float* __restrict__ x_f,
    const float* __restrict__ skipa, float* __restrict__ out)
{
  const int tid = threadIdx.x, lane = tid & 31, wave = tid >> 5;
  const int gm = (int)blockIdx.x * 8 + wave;
  if (gm >= (DSTEPS * BATCH) / 16) return;   // wave-uniform, EXEC stays all-1s
  const float alpha = sigm(skipa[0]);
  const int r0 = gm * 16;
  const int arow = r0 + (lane & 15);
  const int koff = (lane >> 4) << 3;
  const int kb   = (lane >> 4) << 3;

  v8f acc[8];
#pragma unroll
  for (int n = 0; n < 8; ++n)
#pragma unroll
    for (int i = 0; i < 8; ++i) acc[n][i] = 0.f;

  for (int kit = 0; kit < 16; ++kit) {
    const int base = kit * 32 + koff;
    ABu a;
    const bf16_t* ap = yhist + (size_t)arow * HID + base;
    a.h[0] = *(const v8bf*)ap;
    a.h[1] = *(const v8bf*)(ap + 16);
    const int baseB = kit * 32 + kb;
#pragma unroll
    for (int n = 0; n < 8; ++n) {
      const int d = n * 16 + (lane & 15);
      ABu bm;
      const bf16_t* bp = woutb + (size_t)d * HID + baseB;
      bm.h[0] = *(const v8bf*)bp;
      bm.h[1] = *(const v8bf*)(bp + 16);
      acc[n] = wmma_bf16f32(a.v, bm.v, acc[n]);
    }
  }

  const int t  = r0 >> 6;        // rows are t*64 + b, 16 | 64 so t uniform
  const int b0 = r0 & 63;
#pragma unroll
  for (int n = 0; n < 8; ++n) {
    const int d = n * 16 + (lane & 15);
    const float bo = b_out[d];
#pragma unroll
    for (int v = 0; v < 8; ++v) {
      const int b = b0 + v + ((lane >> 4) << 3);
      const float dec = acc[n][v] + bo;
      const float xin = x_f[((size_t)b * TLEN + (t + 1)) * DIN + d];
      out[((size_t)b * DSTEPS + t) * DIN + d] = alpha * dec + (1.f - alpha) * xin;
    }
  }
}

// ---------------------------------------------------------------------------
extern "C" void kernel_launch(void* const* d_in, const int* in_sizes, int n_in,
                              void* d_out, int out_size, void* d_ws, size_t ws_size,
                              hipStream_t stream) {
  (void)in_sizes; (void)n_in; (void)out_size; (void)ws_size;
  const float* x_f   = (const float*)d_in[0];
  const int*   lens  = (const int*)d_in[1];
  const float* Wih_f = (const float*)d_in[2];
  const float* Whh_f = (const float*)d_in[3];
  const float* b_f   = (const float*)d_in[4];
  const float* Wih_b = (const float*)d_in[5];
  const float* Whh_b = (const float*)d_in[6];
  const float* b_b   = (const float*)d_in[7];
  const float* Wih_d = (const float*)d_in[8];
  const float* Whh_d = (const float*)d_in[9];
  const float* b_d   = (const float*)d_in[10];
  const float* W_lat = (const float*)d_in[11];
  const float* b_lat = (const float*)d_in[12];
  const float* W_l2h = (const float*)d_in[13];
  const float* b_l2h = (const float*)d_in[14];
  const float* W_out = (const float*)d_in[15];
  const float* b_out = (const float*)d_in[16];
  const float* skipa = (const float*)d_in[17];

  char* ws = (char*)d_ws;
  size_t off = 0;
  auto take = [&](size_t bytes) -> char* {
    char* p = ws + off;
    off += (bytes + 255) & ~(size_t)255;
    return p;
  };
  bf16_t* xbf     = (bf16_t*)take((size_t)BATCH * TLEN * DIN * 2);
  bf16_t* whh_all = (bf16_t*)take((size_t)3 * G4 * HID * 2);
  bf16_t* wih_all = (bf16_t*)take((size_t)3 * G4 * DIN * 2);
  bf16_t* wout_bf = (bf16_t*)take((size_t)DIN * HID * 2);
  bf16_t* hbuf    = (bf16_t*)take((size_t)2 * 2 * BATCH * HID * 2);
  bf16_t* hdec    = (bf16_t*)take((size_t)BATCH * HID * 2);
  float*  latin   = (float*)take((size_t)BATCH * G4 * 4);
  bf16_t* yhist   = (bf16_t*)take((size_t)DSTEPS * BATCH * HID * 2);
  unsigned int* counters = (unsigned int*)take(256);

  // Re-initialize per call (graph-replay safe): h parity buffers + barriers.
  zero_u32<<<(65536 + 255) / 256, 256, 0, stream>>>((unsigned int*)hbuf, 65536);
  zero_u32<<<1, 64, 0, stream>>>(counters, 64);

  auto cv = [&](bf16_t* dst, const float* src, int n) {
    cvt_kernel<<<(n + 255) / 256, 256, 0, stream>>>(dst, src, n);
  };
  cv(xbf, x_f, BATCH * TLEN * DIN);
  cv(whh_all + (size_t)0 * G4 * HID, Whh_f, G4 * HID);
  cv(whh_all + (size_t)1 * G4 * HID, Whh_b, G4 * HID);
  cv(whh_all + (size_t)2 * G4 * HID, Whh_d, G4 * HID);
  cv(wih_all + (size_t)0 * G4 * DIN, Wih_f, G4 * DIN);
  cv(wih_all + (size_t)1 * G4 * DIN, Wih_b, G4 * DIN);
  cv(wih_all + (size_t)2 * G4 * DIN, Wih_d, G4 * DIN);
  cv(wout_bf, W_out, DIN * HID);

  const int smem = COLS * LDW * 2 + BATCH * GLD * 4 + COLS * 4;  // ~196 KB
  (void)hipFuncSetAttribute(reinterpret_cast<const void*>(lstm_encoder_kernel),
                            hipFuncAttributeMaxDynamicSharedMemorySize, smem);
  (void)hipFuncSetAttribute(reinterpret_cast<const void*>(lstm_decoder_kernel),
                            hipFuncAttributeMaxDynamicSharedMemorySize, smem);

  lstm_encoder_kernel<<<2 * NWG, 256, smem, stream>>>(
      xbf, lens, whh_all, wih_all, b_f, b_b, hbuf, latin, counters);

  latent_kernel<<<1, 256, 0, stream>>>(latin, W_lat, b_lat, W_l2h, b_l2h, hdec);

  lstm_decoder_kernel<<<NWG, 256, smem, stream>>>(
      xbf, whh_all + (size_t)2 * G4 * HID, wih_all + (size_t)2 * G4 * DIN, b_d,
      hdec, yhist, counters + 32);

  out_proj_kernel<<<256, 256, 0, stream>>>(yhist, wout_bf, b_out, x_f, skipa,
                                           (float*)d_out);
}
